// CausalSelfAttention_30872224923937
// MI455X (gfx1250) — matrix operand order repaired and checked by reading, not compiled, not run on hardware
//
#include <hip/hip_runtime.h>
#include <stdint.h>

typedef __attribute__((ext_vector_type(16))) __bf16 v16bf;
typedef __attribute__((ext_vector_type(8)))  __bf16 v8bf;
typedef __attribute__((ext_vector_type(8)))  float  v8f;

#define B_    2
#define T_    2048
#define C_    1024
#define H_    16
#define DH_   64
#define MTOK  (B_ * T_)     // 4096 tokens
#define N3C   (3 * C_)      // 3072

// ---------- helpers ----------

static __device__ __forceinline__ __bf16 f2bf(float f) {
  union { float f; uint32_t u; } v; v.f = f;
  uint32_t r = v.u + 0x7FFFu + ((v.u >> 16) & 1u);   // round-to-nearest-even
  union { unsigned short s; __bf16 b; } o; o.s = (unsigned short)(r >> 16);
  return o.b;
}

static __device__ __forceinline__ v16bf mk16(v8bf lo, v8bf hi) {
  v16bf r;
#pragma unroll
  for (int i = 0; i < 8; ++i) { r[i] = lo[i]; r[i + 8] = hi[i]; }
  return r;
}

static __device__ __forceinline__ v8f wmma_bf16(v16bf a, v16bf b, v8f c) {
  // D = A(16x32 bf16) x B(32x16 bf16) + C(16x16 f32)
  return __builtin_amdgcn_wmma_f32_16x16x32_bf16(
      false, a, false, b, (short)0, c, false, false);
}

static __device__ __forceinline__ float redmax16(float v) {
#pragma unroll
  for (int m = 1; m <= 8; m <<= 1) v = fmaxf(v, __shfl_xor(v, m, 32));
  return v;
}
static __device__ __forceinline__ float redsum16(float v) {
#pragma unroll
  for (int m = 1; m <= 8; m <<= 1) v += __shfl_xor(v, m, 32);
  return v;
}

// A-fragment loader: row-major source, row = tile row + (lane&15)
static __device__ __forceinline__ v16bf loadA(const __bf16* row, int klo) {
  return mk16(*(const v8bf*)(row + klo), *(const v8bf*)(row + klo + 16));
}
// B-fragment loader: transposed-weight source (contiguous over K)
static __device__ __forceinline__ v16bf loadB(const __bf16* col, int kbr) {
  return mk16(*(const v8bf*)(col + kbr), *(const v8bf*)(col + kbr + 16));
}

// ---------- conversion kernels ----------

__global__ __launch_bounds__(256) void cvt_kernel(const float* __restrict__ src,
                                                  __bf16* __restrict__ dst, int n) {
  int i = blockIdx.x * 256 + threadIdx.x;
  if (i < n) dst[i] = f2bf(src[i]);
}

// src[M][N] (f32) -> dst[N][M] (bf16)
__global__ __launch_bounds__(256) void cvt_t_kernel(const float* __restrict__ src,
                                                    __bf16* __restrict__ dst,
                                                    int M, int N) {
  int i = blockIdx.x * 256 + threadIdx.x;
  if (i < M * N) {
    int nn = i / M;
    int mm = i - nn * M;
    dst[i] = f2bf(src[(size_t)mm * N + nn]);
  }
}

// =====================================================================
// Pipelined 32(M) x 64(N) GEMM core, K unrolled by 2 with even/odd
// fragment register sets (no rotation moves -> no WMMA WAR hazard NOPs).
// Declared as a macro-like inline so both GEMM kernels share it.
// =====================================================================
#define GEMM_CORE(arow0, arow1, bcol, acc, klo, kbr)                          \
  do {                                                                        \
    v16bf a0e = loadA((arow0), (klo));                                        \
    v16bf a1e = loadA((arow1), (klo));                                        \
    v16bf be[4], a0o, a1o, bo[4];                                             \
    _Pragma("unroll")                                                         \
    for (int j = 0; j < 4; ++j) be[j] = loadB((bcol)[j], (kbr));              \
    _Pragma("unroll 1")                                                       \
    for (int kk = 0; kk + 64 < C_; kk += 64) {                                \
      /* odd loads (kk+32) fly over even WMMAs (kk) */                        \
      a0o = loadA((arow0) + kk + 32, (klo));                                  \
      a1o = loadA((arow1) + kk + 32, (klo));                                  \
      _Pragma("unroll")                                                       \
      for (int j = 0; j < 4; ++j) bo[j] = loadB((bcol)[j] + kk + 32, (kbr));  \
      _Pragma("unroll")                                                       \
      for (int j = 0; j < 4; ++j) acc[0][j] = wmma_bf16(a0e, be[j], acc[0][j]);\
      _Pragma("unroll")                                                       \
      for (int j = 0; j < 4; ++j) acc[1][j] = wmma_bf16(a1e, be[j], acc[1][j]);\
      /* even loads (kk+64) fly over odd WMMAs (kk+32) */                     \
      a0e = loadA((arow0) + kk + 64, (klo));                                  \
      a1e = loadA((arow1) + kk + 64, (klo));                                  \
      _Pragma("unroll")                                                       \
      for (int j = 0; j < 4; ++j) be[j] = loadB((bcol)[j] + kk + 64, (kbr));  \
      _Pragma("unroll")                                                       \
      for (int j = 0; j < 4; ++j) acc[0][j] = wmma_bf16(a0o, bo[j], acc[0][j]);\
      _Pragma("unroll")                                                       \
      for (int j = 0; j < 4; ++j) acc[1][j] = wmma_bf16(a1o, bo[j], acc[1][j]);\
    }                                                                         \
    /* tail: even frags hold kk = C_-64; odd frags for kk = C_-32 */          \
    a0o = loadA((arow0) + C_ - 32, (klo));                                    \
    a1o = loadA((arow1) + C_ - 32, (klo));                                    \
    _Pragma("unroll")                                                         \
    for (int j = 0; j < 4; ++j) bo[j] = loadB((bcol)[j] + C_ - 32, (kbr));    \
    _Pragma("unroll")                                                         \
    for (int j = 0; j < 4; ++j) acc[0][j] = wmma_bf16(a0e, be[j], acc[0][j]); \
    _Pragma("unroll")                                                         \
    for (int j = 0; j < 4; ++j) acc[1][j] = wmma_bf16(a1e, be[j], acc[1][j]); \
    _Pragma("unroll")                                                         \
    for (int j = 0; j < 4; ++j) acc[0][j] = wmma_bf16(a0o, bo[j], acc[0][j]); \
    _Pragma("unroll")                                                         \
    for (int j = 0; j < 4; ++j) acc[1][j] = wmma_bf16(a1o, bo[j], acc[1][j]); \
  } while (0)

// ---------- QKV projection: qkv = x @ Wqkv + b, scattered into Q,K,[V^T] ----

__global__ __launch_bounds__(256) void qkv_gemm_kernel(
    const __bf16* __restrict__ xb,    // [4096][1024]
    const __bf16* __restrict__ wT,    // [3072][1024]  (Wqkv transposed)
    const float*  __restrict__ bias,  // [3072]
    __bf16* __restrict__ qb,          // [B,H,T,DH]
    __bf16* __restrict__ kb,          // [B,H,T,DH]
    __bf16* __restrict__ vT) {        // [B,H,DH,T]
  const int lane  = threadIdx.x & 31;
  const int wid   = threadIdx.x >> 5;
  const int w     = blockIdx.x * 8 + wid;     // 6144 waves
  const int tn    = w % (N3C / 64);           // 48 N-tiles (64 wide)
  const int tm    = w / (N3C / 64);           // 128 M-tiles (32 tall)
  const int mrow  = lane & 15;
  const int mbase = (lane < 16) ? 0 : 8;
  const int ncol  = lane & 15;
  const int klo   = (lane < 16) ? 0 : 8;      // A-frag K base
  const int kbr   = (lane < 16) ? 0 : 8;      // B-frag K base

  v8f acc[2][4];
#pragma unroll
  for (int i = 0; i < 2; ++i)
#pragma unroll
    for (int j = 0; j < 4; ++j) acc[i][j] = v8f{};

  const __bf16* arow0 = xb + (size_t)(tm * 32 + mrow) * C_;
  const __bf16* arow1 = arow0 + (size_t)16 * C_;
  const __bf16* bcol[4];
#pragma unroll
  for (int j = 0; j < 4; ++j)
    bcol[j] = wT + (size_t)(tn * 64 + j * 16 + ncol) * C_;

  GEMM_CORE(arow0, arow1, bcol, acc, klo, kbr);

  // epilogue: + bias, scatter to Q [B,H,T,DH], K [B,H,T,DH], V^T [B,H,DH,T]
#pragma unroll
  for (int j = 0; j < 4; ++j) {
    const int coln  = tn * 64 + j * 16 + ncol;
    const float bj  = bias[coln];
    const int which = coln >> 10;        // 0=q 1=k 2=v
    const int c     = coln & (C_ - 1);
    const int h     = c >> 6;
    const int d     = c & (DH_ - 1);
#pragma unroll
    for (int i = 0; i < 2; ++i) {
#pragma unroll
      for (int r = 0; r < 8; ++r) {
        const int row = tm * 32 + i * 16 + mbase + r;  // global token
        const int bb  = row >> 11;
        const int t   = row & (T_ - 1);
        const __bf16 val = f2bf(acc[i][j][r] + bj);
        if (which == 2) {
          vT[((size_t)(bb * H_ + h) * DH_ + d) * T_ + t] = val;
        } else {
          __bf16* dst = (which == 0) ? qb : kb;
          dst[((size_t)(bb * H_ + h) * T_ + t) * DH_ + d] = val;
        }
      }
    }
  }
}

// ---------- flash attention: one wave per 16-query tile ----------

__global__ __launch_bounds__(256) void attn_kernel(
    const __bf16* __restrict__ qb, const __bf16* __restrict__ kb,
    const __bf16* __restrict__ vT, __bf16* __restrict__ yb) {
  __shared__ __bf16 plds[8][16 * 32];         // per-wave P staging (1 KB each)

  const int lane  = threadIdx.x & 31;
  const int wid   = threadIdx.x >> 5;
  const int gw    = blockIdx.x * 8 + wid;     // 4096 waves
  const int qt    = gw & 127;                 // q-tile within (b,h)
  const int bh    = gw >> 7;                  // 0..31
  const int bb    = bh >> 4;
  const int h     = bh & 15;
  const int q0    = qt * 16;

  const __bf16* Qp = qb + (size_t)bh * T_ * DH_;
  const __bf16* Kp = kb + (size_t)bh * T_ * DH_;
  const __bf16* Vp = vT + (size_t)bh * DH_ * T_;

  const int mrow  = lane & 15;
  const int mbase = (lane < 16) ? 0 : 8;
  const int ncol  = lane & 15;
  const int klo   = (lane < 16) ? 0 : 8;
  const int kbr   = (lane < 16) ? 0 : 8;

  // Q A-fragments for d=0..31 and d=32..63 (loaded once)
  const __bf16* qr = Qp + (size_t)(q0 + mrow) * DH_;
  v16bf aq0 = loadA(qr, klo);
  v16bf aq1 = loadA(qr + 32, klo);

  float m[8], l[8];
#pragma unroll
  for (int r = 0; r < 8; ++r) { m[r] = -1e30f; l[r] = 0.0f; }
  v8f o[4] = {v8f{}, v8f{}, v8f{}, v8f{}};

  const int nkb = (q0 + 16 + 31) >> 5;        // 32-key blocks, causal bound
  for (int kblk = 0; kblk < nkb; ++kblk) {
    const int k0 = kblk * 32;

    // ---- S = Q K^T (16 x 32): two 16x16 tiles, 4 WMMAs ----
    const __bf16* kr0 = Kp + (size_t)(k0 + ncol) * DH_;
    const __bf16* kr1 = Kp + (size_t)(k0 + 16 + ncol) * DH_;
    v16bf b00 = loadB(kr0, kbr);
    v16bf b01 = loadB(kr0 + 32, kbr);
    v16bf b10 = loadB(kr1, kbr);
    v16bf b11 = loadB(kr1 + 32, kbr);
    v8f s0 = {}, s1 = {};
    s0 = wmma_bf16(aq0, b00, s0);
    s0 = wmma_bf16(aq1, b01, s0);
    s1 = wmma_bf16(aq0, b10, s1);
    s1 = wmma_bf16(aq1, b11, s1);

    // ---- issue V fragment loads now; they overlap the softmax VALU work ----
    v16bf bv[4];
#pragma unroll
    for (int t = 0; t < 4; ++t) {
      const __bf16* vr = Vp + (size_t)(t * 16 + ncol) * T_ + k0 + kbr;
      bv[t] = mk16(*(const v8bf*)(vr), *(const v8bf*)(vr + 16));
    }

    // ---- scale, causal mask, online softmax ----
    float p0[8], p1[8], alpha[8];
#pragma unroll
    for (int r = 0; r < 8; ++r) {
      const int row = q0 + mbase + r;
      float v0 = s0[r] * 0.125f;               // 1/sqrt(64)
      float v1 = s1[r] * 0.125f;
      if (k0 + ncol      > row) v0 = -1e30f;
      if (k0 + 16 + ncol > row) v1 = -1e30f;
      float rmax = redmax16(fmaxf(v0, v1));
      float mn   = fmaxf(m[r], rmax);
      alpha[r]   = __expf(m[r] - mn);
      m[r]       = mn;
      float e0 = __expf(v0 - mn);
      float e1 = __expf(v1 - mn);
      p0[r] = e0; p1[r] = e1;
      l[r] = l[r] * alpha[r] + redsum16(e0 + e1);
    }
#pragma unroll
    for (int t = 0; t < 4; ++t)
#pragma unroll
      for (int r = 0; r < 8; ++r) o[t][r] = o[t][r] * alpha[r];

    // ---- transpose P (C-layout -> A-layout) through LDS ----
    __bf16* pl = plds[wid];
#pragma unroll
    for (int r = 0; r < 8; ++r) {
      pl[(mbase + r) * 32 + ncol]      = f2bf(p0[r]);
      pl[(mbase + r) * 32 + ncol + 16] = f2bf(p1[r]);
    }
    asm volatile("s_wait_dscnt 0" ::: "memory");   // same-wave LDS RAW
    const __bf16* pr = pl + mrow * 32;
    v16bf pa = mk16(*(const v8bf*)(pr + klo), *(const v8bf*)(pr + klo + 16));

    // ---- O += P V : 4 d-tiles, 4 WMMAs (V frags already in flight) ----
#pragma unroll
    for (int t = 0; t < 4; ++t) o[t] = wmma_bf16(pa, bv[t], o[t]);
  }

  // ---- normalize and write y[b, t, h*64 + d] as bf16 ----
  float inv[8];
#pragma unroll
  for (int r = 0; r < 8; ++r) inv[r] = 1.0f / l[r];
#pragma unroll
  for (int t = 0; t < 4; ++t)
#pragma unroll
    for (int r = 0; r < 8; ++r) {
      const int row = q0 + mbase + r;
      yb[((size_t)bb * T_ + row) * C_ + h * DH_ + t * 16 + ncol] =
          f2bf(o[t][r] * inv[r]);
    }
}

// ---------- output projection: out = y @ Wproj + b (fp32 out) ----------

__global__ __launch_bounds__(256) void proj_gemm_kernel(
    const __bf16* __restrict__ yb,    // [4096][1024]
    const __bf16* __restrict__ wT,    // [1024][1024] (Wproj transposed)
    const float*  __restrict__ bias,  // [1024]
    float* __restrict__ out) {        // [4096][1024] f32
  const int lane  = threadIdx.x & 31;
  const int wid   = threadIdx.x >> 5;
  const int w     = blockIdx.x * 8 + wid;     // 2048 waves
  const int tn    = w % (C_ / 64);            // 16 N-tiles
  const int tm    = w / (C_ / 64);            // 128 M-tiles
  const int mrow  = lane & 15;
  const int mbase = (lane < 16) ? 0 : 8;
  const int ncol  = lane & 15;
  const int klo   = (lane < 16) ? 0 : 8;
  const int kbr   = (lane < 16) ? 0 : 8;

  v8f acc[2][4];
#pragma unroll
  for (int i = 0; i < 2; ++i)
#pragma unroll
    for (int j = 0; j < 4; ++j) acc[i][j] = v8f{};

  const __bf16* arow0 = yb + (size_t)(tm * 32 + mrow) * C_;
  const __bf16* arow1 = arow0 + (size_t)16 * C_;
  const __bf16* bcol[4];
#pragma unroll
  for (int j = 0; j < 4; ++j)
    bcol[j] = wT + (size_t)(tn * 64 + j * 16 + ncol) * C_;

  GEMM_CORE(arow0, arow1, bcol, acc, klo, kbr);

#pragma unroll
  for (int j = 0; j < 4; ++j) {
    const int coln = tn * 64 + j * 16 + ncol;
    const float bj = bias[coln];
#pragma unroll
    for (int i = 0; i < 2; ++i) {
#pragma unroll
      for (int r = 0; r < 8; ++r) {
        const int row = tm * 32 + i * 16 + mbase + r;
        out[(size_t)row * C_ + coln] = acc[i][j][r] + bj;
      }
    }
  }
}

// ---------- launch ----------

extern "C" void kernel_launch(void* const* d_in, const int* in_sizes, int n_in,
                              void* d_out, int out_size, void* d_ws, size_t ws_size,
                              hipStream_t stream) {
  const float* x     = (const float*)d_in[0];   // [B,T,C]
  const float* wqkv  = (const float*)d_in[1];   // [C,3C]
  const float* bqkv  = (const float*)d_in[2];   // [3C]
  const float* wproj = (const float*)d_in[3];   // [C,C]
  const float* bproj = (const float*)d_in[4];   // [C]
  float* out = (float*)d_out;

  char* ws = (char*)d_ws;
  const size_t MB = 1024u * 1024u;
  __bf16* xb     = (__bf16*)(ws + 0 * MB);    // 8 MB  [4096,1024]
  __bf16* wqkvT  = (__bf16*)(ws + 8 * MB);    // 6 MB  [3072,1024]
  __bf16* wprojT = (__bf16*)(ws + 14 * MB);   // 2 MB  [1024,1024]
  __bf16* qb     = (__bf16*)(ws + 16 * MB);   // 8 MB  [B,H,T,DH]
  __bf16* kbuf   = (__bf16*)(ws + 24 * MB);   // 8 MB  [B,H,T,DH]
  __bf16* vT     = (__bf16*)(ws + 32 * MB);   // 8 MB  [B,H,DH,T]
  __bf16* yb     = (__bf16*)(ws + 40 * MB);   // 8 MB  [4096,1024]

  cvt_kernel  <<<(MTOK * C_) / 256, 256, 0, stream>>>(x, xb, MTOK * C_);
  cvt_t_kernel<<<(C_ * N3C) / 256, 256, 0, stream>>>(wqkv, wqkvT, C_, N3C);
  cvt_t_kernel<<<(C_ * C_) / 256, 256, 0, stream>>>(wproj, wprojT, C_, C_);

  // 128 x 48 tiles of (32x64), 8 waves/block
  qkv_gemm_kernel<<<(128 * 48) / 8, 256, 0, stream>>>(xb, wqkvT, bqkv, qb, kbuf, vT);
  // 4096 q-tile waves, 8 waves/block
  attn_kernel<<<4096 / 8, 256, 0, stream>>>(qb, kbuf, vT, yb);
  // 128 x 16 tiles of (32x64), 8 waves/block
  proj_gemm_kernel<<<(128 * 16) / 8, 256, 0, stream>>>(yb, wprojT, bproj, out);
}
